// node_tokenizer_15461882265619
// MI455X (gfx1250) — matrix operand order repaired and checked by reading, not compiled, not run on hardware
//
#include <hip/hip_runtime.h>
#include <hip/hip_bf16.h>

// ---------------- problem constants ----------------
#define B_ 128
#define T_ 512
#define D_ 64
#define H_ 512
#define L_ 256
#define G4H (4 * H_)

typedef __bf16 bf16_t;
typedef __attribute__((__ext_vector_type__(16))) __bf16 v16bf;
typedef __attribute__((__ext_vector_type__(8)))  __bf16 v8bf;
typedef __attribute__((__ext_vector_type__(8)))  float  v8f;

// ---------------- WMMA fragment loaders (CDNA5 ISA 7.12.2 layouts) ----------------
// A (16x32 bf16): lane m(0-15) -> row m; elems 0..7 = K sel..sel+7, 8..15 = K sel+16..sel+23
// (sel = 0 lanes 0-15, 8 lanes 16-31). Two 16B global loads per lane.
static __device__ inline v16bf load_a_bf(const bf16_t* __restrict__ base, int rstride,
                                         int rowbase, int kbase, int lane) {
    int m   = lane & 15;
    int sel = (lane >> 4) << 3;
    const bf16_t* p = base + (size_t)(rowbase + m) * rstride + kbase + sel;
    v8bf lo = *(const v8bf*)(p);
    v8bf hi = *(const v8bf*)(p + 16);
    return __builtin_shufflevector(lo, hi, 0,1,2,3,4,5,6,7,8,9,10,11,12,13,14,15);
}

// A from fp32 source with on-the-fly bf16 conversion (same layout)
static __device__ inline v16bf load_a_f32(const float* __restrict__ base, int rstride,
                                          int rowbase, int kbase, int lane) {
    int m   = lane & 15;
    int sel = (lane >> 4) << 3;
    const float* p = base + (size_t)(rowbase + m) * rstride + kbase + sel;
    v16bf a;
#pragma unroll
    for (int e = 0; e < 8; ++e)  a[e] = (bf16_t)p[e];
#pragma unroll
    for (int e = 8; e < 16; ++e) a[e] = (bf16_t)p[e + 8];
    return a;
}

// B (32x16 bf16) from global: lane n -> col n; elems 0..7 = K sel..sel+7, 8..15 = K sel+16..sel+23
// (sel = 0 lanes 0-15, 8 lanes 16-31). Weight row-major (N x K) -> two 16B loads per lane.
static __device__ inline v16bf load_b_bf(const bf16_t* __restrict__ w, int rstride,
                                         int colbase, int kbase, int lane) {
    int n    = lane & 15;
    int koff = (lane >> 4) << 3;
    const bf16_t* p = w + (size_t)(colbase + n) * rstride + kbase + koff;
    v8bf lo = *(const v8bf*)(p);
    v8bf hi = *(const v8bf*)(p + 16);
    return __builtin_shufflevector(lo, hi, 0,1,2,3,4,5,6,7,8,9,10,11,12,13,14,15);
}

static __device__ inline v8f wmma_bf16(v16bf a, v16bf b, v8f c) {
    return __builtin_amdgcn_wmma_f32_16x16x32_bf16(false, a, false, b, (short)0, c,
                                                   false, false);
}

static __device__ inline float sigm(float v) { return 1.0f / (1.0f + __expf(-v)); }

// ------- per-rowgroup step barrier (32 blocks each; sense-free monotonic counter) ------
static __device__ inline void group_sync(unsigned* cnt, unsigned target) {
    __threadfence();                 // flush/order this thread's global stores (agent)
    __syncthreads();
    if (threadIdx.x == 0) {
        __hip_atomic_fetch_add(cnt, 1u, __ATOMIC_RELEASE, __HIP_MEMORY_SCOPE_AGENT);
        while (__hip_atomic_load(cnt, __ATOMIC_ACQUIRE, __HIP_MEMORY_SCOPE_AGENT) < target)
            __builtin_amdgcn_s_sleep(1);
    }
    __syncthreads();
    __builtin_amdgcn_fence(__ATOMIC_ACQUIRE, "agent");  // invalidate L0 for fresh h loads
}

// ---------------- setup kernel ----------------
__global__ __launch_bounds__(256) void cvt_bf16_kernel(const float* __restrict__ in,
                                                       bf16_t* __restrict__ out, int n) {
    int i = blockIdx.x * blockDim.x + threadIdx.x;
    if (i < n) out[i] = (bf16_t)in[i];
}

// ---------------- persistent encoder: all T steps in one launch ----------------
// grid (8,32) = 256 blocks, 128 threads (4 waves; wave w owns gate w's 16x16 tile).
// W_hh/W_ih slices live in VGPRs for the whole kernel; A-side is 4-deep pipelined.
// amdgpu_waves_per_eu(1,4): allow full 256-VGPR budget so the scheduler keeps the
// 4-deep A rotating buffer in flight instead of collapsing to a single buffer.
__global__ __launch_bounds__(128)
__attribute__((amdgpu_waves_per_eu(1, 4)))
void enc_persistent_kernel(
    const float* __restrict__ x, const int* __restrict__ lengths,
    const bf16_t* __restrict__ Wih, const bf16_t* __restrict__ Whh,
    const float* __restrict__ be,
    bf16_t* __restrict__ h0, bf16_t* __restrict__ h1,
    float* __restrict__ hfp, float* __restrict__ cfp, unsigned* __restrict__ cnt) {
    __shared__ float gs[4][16][16];
    __shared__ float bes[4][16];
    __shared__ int   lenlds[16];

    const int lane    = threadIdx.x & 31;
    const int wv      = threadIdx.x >> 5;       // gate 0..3
    const int rowbase = blockIdx.x * 16;        // batch tile
    const int hbase   = blockIdx.y * 16;        // hidden tile
    const int gcol    = wv * H_ + hbase;        // column base within 4H gate dim
    unsigned* my_cnt  = cnt + blockIdx.x * 64;  // 256B-spaced counter per row-group
    const unsigned grp = gridDim.y;             // 32 blocks per row-group

    // ---- one-time: weight fragments into registers (t-invariant B operands) ----
    v16bf bw[16];
#pragma unroll
    for (int i = 0; i < 16; ++i) bw[i] = load_b_bf(Whh, H_, gcol, i * 32, lane);
    v16bf bx0 = load_b_bf(Wih, D_, gcol, 0,  lane);
    v16bf bx1 = load_b_bf(Wih, D_, gcol, 32, lane);

    if (threadIdx.x < 64) bes[threadIdx.x >> 4][threadIdx.x & 15] =
        be[(threadIdx.x >> 4) * H_ + hbase + (threadIdx.x & 15)];
    if (threadIdx.x < 16) lenlds[threadIdx.x] = lengths[rowbase + threadIdx.x];
    __syncthreads();

    for (int t = 0; t < T_; ++t) {
        const bf16_t* hin  = (t & 1) ? h1 : h0;
        bf16_t*       hout = (t & 1) ? h0 : h1;
        const float*  xrow = x + (size_t)t * D_;

        // issue x fragments early: their loads+cvt overlap the 16 recurrent WMMAs
        v16bf ax0 = load_a_f32(xrow, T_ * D_, rowbase, 0,  lane);
        v16bf ax1 = load_a_f32(xrow, T_ * D_, rowbase, 32, lane);

        // 4-deep software pipeline on the A (h_prev) fragments
        v16bf afr[4];
#pragma unroll
        for (int i = 0; i < 4; ++i) afr[i] = load_a_bf(hin, H_, rowbase, i * 32, lane);

        v8f acc = {0.f, 0.f, 0.f, 0.f, 0.f, 0.f, 0.f, 0.f};
#pragma unroll
        for (int i = 0; i < 16; ++i) {
            v16bf cur = afr[i & 3];
            if (i + 4 < 16) afr[i & 3] = load_a_bf(hin, H_, rowbase, (i + 4) * 32, lane);
            acc = wmma_bf16(cur, bw[i], acc);
        }
        acc = wmma_bf16(ax0, bx0, acc);
        acc = wmma_bf16(ax1, bx1, acc);

        // gate tile -> LDS (C layout: elem r -> row r + 8*(lane>=16), col lane&15)
#pragma unroll
        for (int r = 0; r < 8; ++r)
            gs[wv][r + ((lane >> 4) << 3)][lane & 15] = acc[r];
        __syncthreads();

        // LSTM cell update, fp32 master state
        for (int e = threadIdx.x; e < 256; e += 128) {
            int m = e >> 4, n = e & 15;
            size_t idx = (size_t)(rowbase + m) * H_ + hbase + n;
            float gi = gs[0][m][n] + bes[0][n];
            float gf = gs[1][m][n] + bes[1][n];
            float gg = gs[2][m][n] + bes[2][n];
            float go = gs[3][m][n] + bes[3][n];
            float i = sigm(gi), f = sigm(gf), g = tanhf(gg), o = sigm(go);
            float c_old = cfp[idx];
            float c_new = f * c_old + i * g;
            float h_new = o * tanhf(c_new);
            bool  mk = (t < lenlds[m]);
            float h2 = mk ? h_new : hfp[idx];
            float c2 = mk ? c_new : c_old;
            hfp[idx]  = h2;
            cfp[idx]  = c2;
            hout[idx] = (bf16_t)h2;
        }
        group_sync(my_cnt, (unsigned)(t + 1) * grp);
    }
}

// ---------------- persistent decoder ----------------
__global__ __launch_bounds__(128)
__attribute__((amdgpu_waves_per_eu(1, 4)))
void dec_persistent_kernel(
    const int* __restrict__ lengths, const float* __restrict__ zx,
    const bf16_t* __restrict__ Whh,
    bf16_t* __restrict__ h0, bf16_t* __restrict__ h1,
    float* __restrict__ hfp, float* __restrict__ cfp,
    bf16_t* __restrict__ ys, unsigned* __restrict__ cnt) {
    __shared__ float gs[4][16][16];
    __shared__ float zxlds[4][16][16];
    __shared__ int   lenlds[16];

    const int lane    = threadIdx.x & 31;
    const int wv      = threadIdx.x >> 5;
    const int rowbase = blockIdx.x * 16;
    const int hbase   = blockIdx.y * 16;
    const int gcol    = wv * H_ + hbase;
    unsigned* my_cnt  = cnt + blockIdx.x * 64;
    const unsigned grp = gridDim.y;

    v16bf bw[16];
#pragma unroll
    for (int i = 0; i < 16; ++i) bw[i] = load_b_bf(Whh, H_, gcol, i * 32, lane);

    for (int idx = threadIdx.x; idx < 4 * 16 * 16; idx += 128) {
        int g = idx >> 8, m = (idx >> 4) & 15, n = idx & 15;   // zx already contains b_d
        zxlds[g][m][n] = zx[(size_t)(rowbase + m) * G4H + g * H_ + hbase + n];
    }
    if (threadIdx.x < 16) lenlds[threadIdx.x] = lengths[rowbase + threadIdx.x];
    __syncthreads();

    for (int t = 0; t < T_; ++t) {
        const bf16_t* hin  = (t & 1) ? h1 : h0;
        bf16_t*       hout = (t & 1) ? h0 : h1;

        v16bf afr[4];
#pragma unroll
        for (int i = 0; i < 4; ++i) afr[i] = load_a_bf(hin, H_, rowbase, i * 32, lane);

        v8f acc = {0.f, 0.f, 0.f, 0.f, 0.f, 0.f, 0.f, 0.f};
#pragma unroll
        for (int i = 0; i < 16; ++i) {
            v16bf cur = afr[i & 3];
            if (i + 4 < 16) afr[i & 3] = load_a_bf(hin, H_, rowbase, (i + 4) * 32, lane);
            acc = wmma_bf16(cur, bw[i], acc);
        }
#pragma unroll
        for (int r = 0; r < 8; ++r)
            gs[wv][r + ((lane >> 4) << 3)][lane & 15] = acc[r];
        __syncthreads();

        for (int e = threadIdx.x; e < 256; e += 128) {
            int m = e >> 4, n = e & 15;
            int b = rowbase + m, h = hbase + n;
            size_t idx = (size_t)b * H_ + h;
            float gi = gs[0][m][n] + zxlds[0][m][n];
            float gf = gs[1][m][n] + zxlds[1][m][n];
            float gg = gs[2][m][n] + zxlds[2][m][n];
            float go = gs[3][m][n] + zxlds[3][m][n];
            float i = sigm(gi), f = sigm(gf), g = tanhf(gg), o = sigm(go);
            float c_old = cfp[idx];
            float c_new = f * c_old + i * g;
            float h_new = o * tanhf(c_new);
            bool  mk = (t < lenlds[m]);
            float h2 = mk ? h_new : hfp[idx];
            float c2 = mk ? c_new : c_old;
            hfp[idx]  = h2;
            cfp[idx]  = c2;
            hout[idx] = (bf16_t)h2;
            ys[((size_t)t * B_ + b) * H_ + h] = mk ? (bf16_t)h_new : (bf16_t)0.0f;
        }
        group_sync(my_cnt, (unsigned)(t + 1) * grp);
    }
}

// ---------------- latent projections (tiny FLOPs; scalar fp32) ----------------
__global__ __launch_bounds__(256) void latent_z_kernel(const float* __restrict__ h,
                                                       const float* __restrict__ Wlat,
                                                       const float* __restrict__ blat,
                                                       float* __restrict__ z) {
    int gid = blockIdx.x * blockDim.x + threadIdx.x;   // B*L
    int b = gid >> 8, l = gid & (L_ - 1);
    const float* hr = h + (size_t)b * H_;
    const float* wr = Wlat + (size_t)l * H_;
    float s = blat[l];
    for (int k = 0; k < H_; ++k) s = fmaf(hr[k], wr[k], s);
    z[gid] = s;
}

__global__ __launch_bounds__(256) void latent_zx_kernel(const float* __restrict__ z,
                                                        const float* __restrict__ Wihd,
                                                        const float* __restrict__ bd,
                                                        float* __restrict__ zx) {
    int gid = blockIdx.x * blockDim.x + threadIdx.x;   // B*4H
    int b = gid >> 11, g = gid & (G4H - 1);
    const float* zr = z + (size_t)b * L_;
    const float* wr = Wihd + (size_t)g * L_;
    float s = bd[g];
    for (int k = 0; k < L_; ++k) s = fmaf(zr[k], wr[k], s);
    zx[gid] = s;
}

// ---------------- output projection: out[b,t,d] = ys[t,b,:] . W_out[d,:] + b_out ----------
__global__ __launch_bounds__(128) void out_proj_kernel(const bf16_t* __restrict__ ys,
                                                       const bf16_t* __restrict__ Wout,
                                                       const float* __restrict__ bout,
                                                       float* __restrict__ out) {
    const int lane = threadIdx.x & 31;
    const int wv   = threadIdx.x >> 5;
    const int rowbase = blockIdx.x * 16;
    const int dbase   = wv * 16;

    v8f acc = {0.f, 0.f, 0.f, 0.f, 0.f, 0.f, 0.f, 0.f};
#pragma unroll 4
    for (int kb = 0; kb < H_; kb += 32) {
        v16bf a = load_a_bf(ys, H_, rowbase, kb, lane);
        v16bf b = load_b_bf(Wout, H_, dbase, kb, lane);
        acc = wmma_bf16(a, b, acc);
    }
    int d = dbase + (lane & 15);
    float bias = bout[d];
#pragma unroll
    for (int r = 0; r < 8; ++r) {
        int tb = rowbase + r + ((lane >> 4) << 3);
        int t = tb >> 7, b = tb & (B_ - 1);        // B = 128
        out[((size_t)b * T_ + t) * D_ + d] = acc[r] + bias;
    }
}

// ---------------- host launcher ----------------
extern "C" void kernel_launch(void* const* d_in, const int* in_sizes, int n_in,
                              void* d_out, int out_size, void* d_ws, size_t ws_size,
                              hipStream_t stream) {
    (void)in_sizes; (void)n_in; (void)out_size; (void)ws_size;
    const float* x      = (const float*)d_in[0];
    const int*   lens   = (const int*)d_in[1];
    const float* W_ih_e = (const float*)d_in[2];
    const float* W_hh_e = (const float*)d_in[3];
    const float* b_e    = (const float*)d_in[4];
    const float* W_lat  = (const float*)d_in[5];
    const float* b_lat  = (const float*)d_in[6];
    const float* W_ih_d = (const float*)d_in[7];
    const float* W_hh_d = (const float*)d_in[8];
    const float* b_d    = (const float*)d_in[9];
    const float* W_out  = (const float*)d_in[10];
    const float* b_out  = (const float*)d_in[11];
    float* out = (float*)d_out;

    // ---- workspace carve-out (256B aligned) ----
    char* base = (char*)d_ws;
    size_t off = 0;
    auto take = [&](size_t bytes) -> void* {
        void* p = base + off;
        off += (bytes + 255) & ~(size_t)255;
        return p;
    };
    bf16_t*   Wih_e_bf = (bf16_t*)take((size_t)G4H * D_ * 2);
    bf16_t*   Whh_e_bf = (bf16_t*)take((size_t)G4H * H_ * 2);
    bf16_t*   Whh_d_bf = (bf16_t*)take((size_t)G4H * H_ * 2);
    bf16_t*   Wout_bf  = (bf16_t*)take((size_t)D_ * H_ * 2);
    bf16_t*   hbf_e0   = (bf16_t*)take((size_t)B_ * H_ * 2);
    bf16_t*   hbf_e1   = (bf16_t*)take((size_t)B_ * H_ * 2);
    float*    hfp_e    = (float*)take((size_t)B_ * H_ * 4);
    float*    c_e      = (float*)take((size_t)B_ * H_ * 4);
    bf16_t*   hbf_d0   = (bf16_t*)take((size_t)B_ * H_ * 2);
    bf16_t*   hbf_d1   = (bf16_t*)take((size_t)B_ * H_ * 2);
    float*    hfp_d    = (float*)take((size_t)B_ * H_ * 4);
    float*    c_d      = (float*)take((size_t)B_ * H_ * 4);
    float*    z        = (float*)take((size_t)B_ * L_ * 4);
    float*    zx       = (float*)take((size_t)B_ * G4H * 4);
    unsigned* cnt_e    = (unsigned*)take(8 * 64 * sizeof(unsigned));  // 8 row-group counters
    unsigned* cnt_d    = (unsigned*)take(8 * 64 * sizeof(unsigned));
    bf16_t*   ys       = (bf16_t*)take((size_t)T_ * B_ * H_ * 2);

    // ---- one-time bf16 weight conversion ----
    auto cvt = [&](const float* src, bf16_t* dst, int n) {
        cvt_bf16_kernel<<<(n + 255) / 256, 256, 0, stream>>>(src, dst, n);
    };
    cvt(W_ih_e, Wih_e_bf, G4H * D_);
    cvt(W_hh_e, Whh_e_bf, G4H * H_);
    cvt(W_hh_d, Whh_d_bf, G4H * H_);
    cvt(W_out,  Wout_bf,  D_ * H_);

    // ---- zero initial state + barrier counters (replayed every graph iteration) ----
    hipMemsetAsync(hfp_e,  0, (size_t)B_ * H_ * 4, stream);
    hipMemsetAsync(c_e,    0, (size_t)B_ * H_ * 4, stream);
    hipMemsetAsync(hbf_e0, 0, (size_t)B_ * H_ * 2, stream);
    hipMemsetAsync(hfp_d,  0, (size_t)B_ * H_ * 4, stream);
    hipMemsetAsync(c_d,    0, (size_t)B_ * H_ * 4, stream);
    hipMemsetAsync(hbf_d0, 0, (size_t)B_ * H_ * 2, stream);
    hipMemsetAsync(cnt_e,  0, 8 * 64 * sizeof(unsigned), stream);
    hipMemsetAsync(cnt_d,  0, 8 * 64 * sizeof(unsigned), stream);

    dim3 step_grid(B_ / 16, H_ / 16);   // (8, 32) = 256 co-resident blocks

    // ---- encoder: single persistent launch, 512 row-group-synced steps ----
    enc_persistent_kernel<<<step_grid, 128, 0, stream>>>(
        x, lens, Wih_e_bf, Whh_e_bf, b_e, hbf_e0, hbf_e1, hfp_e, c_e, cnt_e);

    // ---- latent projections ----
    latent_z_kernel<<<(B_ * L_) / 256, 256, 0, stream>>>(hfp_e, W_lat, b_lat, z);
    latent_zx_kernel<<<(B_ * G4H) / 256, 256, 0, stream>>>(z, W_ih_d, b_d, zx);

    // ---- decoder: single persistent launch ----
    dec_persistent_kernel<<<step_grid, 128, 0, stream>>>(
        lens, zx, Whh_d_bf, hbf_d0, hbf_d1, hfp_d, c_d, ys, cnt_d);

    // ---- output projection (WMMA GEMM over all T*B rows) ----
    out_proj_kernel<<<(T_ * B_) / 16, 128, 0, stream>>>(ys, Wout_bf, b_out, out);
}
